// Encoder_7413113553686
// MI455X (gfx1250) — matrix operand order repaired and checked by reading, not compiled, not run on hardware
//
#include <hip/hip_runtime.h>

// ---------------------------------------------------------------------------
// MI455X / gfx1250, wave32. D = A(16x32,f16) * B(32x16,f16) + C(16x16,f32)
// via v_wmma_f32_16x16x32_f16.
//
// Only layer L-1 of the reference contributes to the output (each block reads
// the same x0 and overwrites `out`), so we compute exactly one attention layer.
//
// v2: 32x64 register-blocked GEMMs (8 WMMA / k-step per wave), 32-query
//     attention tiles (24 WMMA / 32-key block, K/V fragments reused 2x),
//     global_prefetch_b8 ahead of the GEMM streams.
// ---------------------------------------------------------------------------

typedef __attribute__((ext_vector_type(16))) _Float16 v16h;
typedef __attribute__((ext_vector_type(8)))  _Float16 v8h;
typedef __attribute__((ext_vector_type(8)))  float    v8f;

#define NLAY_  21
#define SCALE_ 0.10206207261596575f   // 96^-0.5

// A fragment (16x32 f16, M x K): lane m = l&15, h = l>>4.
// element e<8  -> k = kbase + 8h + e ; e>=8 -> k = kbase + 16 + 8h + (e-8)
__device__ __forceinline__ v16h ld_frag_a(const _Float16* p, int hh) {
  const _Float16* q = p + hh * 8;
  v8h lo = *(const v8h*)(q);
  v8h hi = *(const v8h*)(q + 16);
  v16h r;
#pragma unroll
  for (int i = 0; i < 8; ++i) { r[i] = lo[i]; r[i + 8] = hi[i]; }
  return r;
}

// B fragment (32x16 f16, K x N): lane n = l&15, h = l>>4.
// element e<8  -> k = kbase + 8h + e ; e>=8 -> k = kbase + 16 + 8h + (e-8)
__device__ __forceinline__ v16h ld_frag_b(const _Float16* p, int hh) {
  const _Float16* q = p + hh * 8;
  v8h lo = *(const v8h*)(q);
  v8h hi = *(const v8h*)(q + 16);
  v16h r;
#pragma unroll
  for (int i = 0; i < 8; ++i) { r[i] = lo[i]; r[i + 8] = hi[i]; }
  return r;
}

__device__ __forceinline__ v8f wmma32(v16h a, v16h b, v8f c) {
  return __builtin_amdgcn_wmma_f32_16x16x32_f16(false, a, false, b,
                                                (short)0, c, false, false);
}

// ---------------------------------------------------------------------------
// Kernel A: cast + transpose layer-20 weights into one f16 matrix
//   wT[row][c], row 0..767 = qw[:,row], 768..2303 = kvw[:,row-768],
//   2304..3071 = pw[:,row-2304].  K (=c) contiguous per row -> b128 B loads.
// ---------------------------------------------------------------------------
__global__ void wconv_kernel(const float* __restrict__ qw,
                             const float* __restrict__ kvw,
                             const float* __restrict__ pw,
                             _Float16* __restrict__ wT) {
  int i = blockIdx.x * blockDim.x + threadIdx.x;
  const int total = 3072 * 768;
  if (i >= total) return;
  int row = i / 768, c = i - row * 768;
  float v;
  if (row < 768)       v = qw[(size_t)c * 768 + row];
  else if (row < 2304) v = kvw[(size_t)c * 1536 + (row - 768)];
  else                 v = pw[(size_t)c * 768 + (row - 2304)];
  wT[i] = (_Float16)v;
}

// ---------------------------------------------------------------------------
// Kernel B: patch-embed conv (8x8, stride 8, CIN=1) + LayerNorm + pos
//   one 256-thread block per token; each thread owns 3 of 768 channels.
// ---------------------------------------------------------------------------
__global__ void patch_ln_kernel(const float* __restrict__ sam,
                                const float* __restrict__ cw,
                                const float* __restrict__ cb,
                                const float* __restrict__ g,
                                const float* __restrict__ bb,
                                const float* __restrict__ pos,
                                float* __restrict__ x0,
                                _Float16* __restrict__ x0h) {
  __shared__ float patch[64];
  __shared__ float red[256], red2[256];
  int tok = blockIdx.x;
  int t = threadIdx.x;
  int b = tok >> 10, n = tok & 1023;
  int gy = n >> 5, gx = n & 31;
  if (t < 64) {
    int py = t >> 3, px = t & 7;
    patch[t] = sam[((size_t)(b * 256) + gy * 8 + py) * 256 + gx * 8 + px];
  }
  __syncthreads();
  float yv[3];
  float s = 0.f, s2 = 0.f;
#pragma unroll
  for (int j = 0; j < 3; ++j) {
    int d = t + j * 256;
    float acc = cb[d];
    const float* w = cw + (size_t)d * 64;
#pragma unroll 8
    for (int i = 0; i < 64; ++i) acc += patch[i] * w[i];
    yv[j] = acc; s += acc; s2 += acc * acc;
  }
  red[t] = s; red2[t] = s2;
  __syncthreads();
  for (int off = 128; off > 0; off >>= 1) {
    if (t < off) { red[t] += red[t + off]; red2[t] += red2[t + off]; }
    __syncthreads();
  }
  float mu  = red[0] * (1.f / 768.f);
  float var = red2[0] * (1.f / 768.f) - mu * mu;
  float rstd = rsqrtf(var + 1e-5f);
#pragma unroll
  for (int j = 0; j < 3; ++j) {
    int d = t + j * 256;
    float v = (yv[j] - mu) * rstd * g[d] + bb[d] + pos[(size_t)n * 768 + d];
    x0 [(size_t)tok * 768 + d] = v;
    x0h[(size_t)tok * 768 + d] = (_Float16)v;
  }
}

// ---------------------------------------------------------------------------
// Kernel C: fused QKV GEMM.  [4096x768] x [768x2304] f16, f32 accum.
//   32x64 tile per wave: 2 A frags x 4 B frags -> 8 WMMAs per 32-wide k-step.
//   Q scaled by 96^-0.5 at writeout; V stored transposed [B,H,96,N].
// ---------------------------------------------------------------------------
__global__ void qkv_gemm_kernel(const _Float16* __restrict__ x0h,
                                const _Float16* __restrict__ wT,
                                _Float16* __restrict__ qs,
                                _Float16* __restrict__ ks,
                                _Float16* __restrict__ vt) {
  int lane = threadIdx.x & 31;
  int wid  = threadIdx.x >> 5;              // 0..3
  int m = lane & 15, hh = lane >> 4;
  int row0 = blockIdx.x * 32;
  int c0   = (blockIdx.y * 4 + wid) * 64;   // col group 0..35 -> cols 0..2303
  const _Float16* ap0 = x0h + (size_t)(row0 +      m) * 768;
  const _Float16* ap1 = x0h + (size_t)(row0 + 16 + m) * 768;
  const _Float16* bp0 = wT + (size_t)(c0 +      m) * 768;
  const _Float16* bp1 = wT + (size_t)(c0 + 16 + m) * 768;
  const _Float16* bp2 = wT + (size_t)(c0 + 32 + m) * 768;
  const _Float16* bp3 = wT + (size_t)(c0 + 48 + m) * 768;
  v8f zero = {};
  v8f acc[2][4];
#pragma unroll
  for (int i = 0; i < 2; ++i)
#pragma unroll
    for (int j = 0; j < 4; ++j) acc[i][j] = zero;
#pragma unroll 2
  for (int k = 0; k < 768; k += 32) {
    __builtin_prefetch(ap0 + k + 128, 0, 3);   // -> global_prefetch_b8
    __builtin_prefetch(bp0 + k + 128, 0, 3);
    v16h a0 = ld_frag_a(ap0 + k, hh);
    v16h a1 = ld_frag_a(ap1 + k, hh);
    v16h b0 = ld_frag_b(bp0 + k, hh);
    v16h b1 = ld_frag_b(bp1 + k, hh);
    v16h b2 = ld_frag_b(bp2 + k, hh);
    v16h b3 = ld_frag_b(bp3 + k, hh);
    acc[0][0] = wmma32(a0, b0, acc[0][0]);
    acc[1][0] = wmma32(a1, b0, acc[1][0]);
    acc[0][1] = wmma32(a0, b1, acc[0][1]);
    acc[1][1] = wmma32(a1, b1, acc[1][1]);
    acc[0][2] = wmma32(a0, b2, acc[0][2]);
    acc[1][2] = wmma32(a1, b2, acc[1][2]);
    acc[0][3] = wmma32(a0, b3, acc[0][3]);
    acc[1][3] = wmma32(a1, b3, acc[1][3]);
  }
#pragma unroll
  for (int i = 0; i < 2; ++i) {
#pragma unroll
    for (int j = 0; j < 4; ++j) {
      int cg = c0 + 16 * j + m;              // this lane's output column
#pragma unroll
      for (int r = 0; r < 8; ++r) {
        int tok = row0 + 16 * i + r + 8 * hh;
        int b = tok >> 10, nn = tok & 1023;
        float v = acc[i][j][r];
        if (cg < 768) {
          int hq = cg / 96, hd = cg % 96;
          qs[((size_t)(b * 8 + hq) * 1024 + nn) * 96 + hd] =
              (_Float16)(v * SCALE_);
        } else if (cg < 1536) {
          int d = cg - 768; int hq = d / 96, hd = d % 96;
          ks[((size_t)(b * 8 + hq) * 1024 + nn) * 96 + hd] = (_Float16)v;
        } else {
          int d = cg - 1536; int hq = d / 96, hd = d % 96;
          vt[((size_t)(b * 8 + hq) * 96 + hd) * 1024 + nn] = (_Float16)v;
        }
      }
    }
  }
}

// ---------------------------------------------------------------------------
// Kernel D: flash attention, one wave per 32-query tile, key blocks of 32.
//   Per key block: 12 WMMAs (QK^T, K frags reused across 2 query sub-tiles),
//   online softmax (shfl-xor reductions in 16-lane halves, v_exp_f32),
//   P staged via LDS (C-layout -> A-layout), 12 WMMAs (P.V, V frags reused).
// ---------------------------------------------------------------------------
__global__ void attn_kernel(const _Float16* __restrict__ qs,
                            const _Float16* __restrict__ ks,
                            const _Float16* __restrict__ vt,
                            _Float16* __restrict__ oh) {
  __shared__ _Float16 sP[4][2 * 16 * 32];
  int lane = threadIdx.x & 31;
  int wid  = threadIdx.x >> 5;
  int m = lane & 15, hh = lane >> 4;
  int hq = blockIdx.y;
  int b  = blockIdx.z;
  int q0 = (blockIdx.x * 4 + wid) * 32;
  const _Float16* Qp = qs + (size_t)(b * 8 + hq) * 1024 * 96;
  const _Float16* Kp = ks + (size_t)(b * 8 + hq) * 1024 * 96;
  const _Float16* Vp = vt + (size_t)(b * 8 + hq) * 96 * 1024;

  v16h aq[2][3];
#pragma unroll
  for (int t = 0; t < 2; ++t)
#pragma unroll
    for (int c = 0; c < 3; ++c)
      aq[t][c] = ld_frag_a(Qp + (size_t)(q0 + 16 * t + m) * 96 + c * 32, hh);

  float mrow[2][8], lrow[2][8];
#pragma unroll
  for (int t = 0; t < 2; ++t)
#pragma unroll
    for (int r = 0; r < 8; ++r) { mrow[t][r] = -3.0e38f; lrow[t][r] = 0.f; }
  v8f zero = {};
  v8f oacc[2][6];
#pragma unroll
  for (int t = 0; t < 2; ++t)
#pragma unroll
    for (int f = 0; f < 6; ++f) oacc[t][f] = zero;

  _Float16* pbuf = &sP[wid][0];

  for (int key0 = 0; key0 < 1024; key0 += 32) {
    v8f s[2][2];
    s[0][0] = zero; s[0][1] = zero; s[1][0] = zero; s[1][1] = zero;
#pragma unroll
    for (int c = 0; c < 3; ++c) {
      v16h bk0 = ld_frag_b(Kp + (size_t)(key0 +      m) * 96 + c * 32, hh);
      v16h bk1 = ld_frag_b(Kp + (size_t)(key0 + 16 + m) * 96 + c * 32, hh);
#pragma unroll
      for (int t = 0; t < 2; ++t) {
        s[t][0] = wmma32(aq[t][c], bk0, s[t][0]);
        s[t][1] = wmma32(aq[t][c], bk1, s[t][1]);
      }
    }
#pragma unroll
    for (int t = 0; t < 2; ++t) {
#pragma unroll
      for (int r = 0; r < 8; ++r) {
        float mx = fmaxf(s[t][0][r], s[t][1][r]);
        mx = fmaxf(mx, __shfl_xor(mx, 1, 32));
        mx = fmaxf(mx, __shfl_xor(mx, 2, 32));
        mx = fmaxf(mx, __shfl_xor(mx, 4, 32));
        mx = fmaxf(mx, __shfl_xor(mx, 8, 32));
        float mnew = fmaxf(mrow[t][r], mx);
        float corr = __expf(mrow[t][r] - mnew);
        float p0 = __expf(s[t][0][r] - mnew);
        float p1 = __expf(s[t][1][r] - mnew);
        float ps = p0 + p1;
        ps += __shfl_xor(ps, 1, 32);
        ps += __shfl_xor(ps, 2, 32);
        ps += __shfl_xor(ps, 4, 32);
        ps += __shfl_xor(ps, 8, 32);
        lrow[t][r] = lrow[t][r] * corr + ps;
        mrow[t][r] = mnew;
#pragma unroll
        for (int f = 0; f < 6; ++f) oacc[t][f][r] = oacc[t][f][r] * corr;
        int prow = r + 8 * hh;               // C layout: lane = key column
        pbuf[t * 512 + prow * 32 + m]      = (_Float16)p0;
        pbuf[t * 512 + prow * 32 + 16 + m] = (_Float16)p1;
      }
    }
    // Re-read P as A fragments (16 queries x 32 keys each) from LDS.
    v16h ap0 = ld_frag_a(pbuf +       m * 32, hh);
    v16h ap1 = ld_frag_a(pbuf + 512 + m * 32, hh);
#pragma unroll
    for (int f = 0; f < 6; ++f) {
      v16h bv = ld_frag_b(Vp + (size_t)(f * 16 + m) * 1024 + key0, hh);
      oacc[0][f] = wmma32(ap0, bv, oacc[0][f]);
      oacc[1][f] = wmma32(ap1, bv, oacc[1][f]);
    }
  }
#pragma unroll
  for (int t = 0; t < 2; ++t) {
#pragma unroll
    for (int r = 0; r < 8; ++r) {
      float inv = 1.f / lrow[t][r];
      int tok = b * 1024 + q0 + 16 * t + r + 8 * hh;
#pragma unroll
      for (int f = 0; f < 6; ++f)
        oh[(size_t)tok * 768 + hq * 96 + f * 16 + m] =
            (_Float16)(oacc[t][f][r] * inv);
    }
  }
}

// ---------------------------------------------------------------------------
// Kernel E: projection GEMM + bias + residual, fp32 output. 32x64 per wave.
// ---------------------------------------------------------------------------
__global__ void proj_kernel(const _Float16* __restrict__ oh,
                            const _Float16* __restrict__ wT,
                            const float* __restrict__ pb,
                            const float* __restrict__ x0,
                            float* __restrict__ out) {
  int lane = threadIdx.x & 31;
  int wid  = threadIdx.x >> 5;              // 0..3
  int m = lane & 15, hh = lane >> 4;
  int row0 = blockIdx.x * 32;
  int c0   = (blockIdx.y * 4 + wid) * 64;   // 0..704
  const _Float16* ap0 = oh + (size_t)(row0 +      m) * 768;
  const _Float16* ap1 = oh + (size_t)(row0 + 16 + m) * 768;
  const _Float16* bp0 = wT + (size_t)(2304 + c0 +      m) * 768;
  const _Float16* bp1 = wT + (size_t)(2304 + c0 + 16 + m) * 768;
  const _Float16* bp2 = wT + (size_t)(2304 + c0 + 32 + m) * 768;
  const _Float16* bp3 = wT + (size_t)(2304 + c0 + 48 + m) * 768;
  v8f zero = {};
  v8f acc[2][4];
#pragma unroll
  for (int i = 0; i < 2; ++i)
#pragma unroll
    for (int j = 0; j < 4; ++j) acc[i][j] = zero;
#pragma unroll 2
  for (int k = 0; k < 768; k += 32) {
    __builtin_prefetch(ap0 + k + 128, 0, 3);
    __builtin_prefetch(bp0 + k + 128, 0, 3);
    v16h a0 = ld_frag_a(ap0 + k, hh);
    v16h a1 = ld_frag_a(ap1 + k, hh);
    v16h b0 = ld_frag_b(bp0 + k, hh);
    v16h b1 = ld_frag_b(bp1 + k, hh);
    v16h b2 = ld_frag_b(bp2 + k, hh);
    v16h b3 = ld_frag_b(bp3 + k, hh);
    acc[0][0] = wmma32(a0, b0, acc[0][0]);
    acc[1][0] = wmma32(a1, b0, acc[1][0]);
    acc[0][1] = wmma32(a0, b1, acc[0][1]);
    acc[1][1] = wmma32(a1, b1, acc[1][1]);
    acc[0][2] = wmma32(a0, b2, acc[0][2]);
    acc[1][2] = wmma32(a1, b2, acc[1][2]);
    acc[0][3] = wmma32(a0, b3, acc[0][3]);
    acc[1][3] = wmma32(a1, b3, acc[1][3]);
  }
#pragma unroll
  for (int i = 0; i < 2; ++i) {
#pragma unroll
    for (int j = 0; j < 4; ++j) {
      int cg = c0 + 16 * j + m;
      float bias = pb[cg];
#pragma unroll
      for (int r = 0; r < 8; ++r) {
        int tok = row0 + 16 * i + r + 8 * hh;
        size_t idx = (size_t)tok * 768 + cg;
        out[idx] = acc[i][j][r] + bias + x0[idx];
      }
    }
  }
}

// ---------------------------------------------------------------------------
// Host: workspace layout (42,467,328 bytes total; oh aliases dead x0h).
// ---------------------------------------------------------------------------
extern "C" void kernel_launch(void* const* d_in, const int* in_sizes, int n_in,
                              void* d_out, int out_size, void* d_ws, size_t ws_size,
                              hipStream_t stream) {
  (void)in_sizes; (void)n_in; (void)out_size; (void)ws_size;
  const float* sam   = (const float*)d_in[0];
  const float* convw = (const float*)d_in[1];
  const float* convb = (const float*)d_in[2];
  const float* lng   = (const float*)d_in[3];
  const float* lnb   = (const float*)d_in[4];
  const float* pos   = (const float*)d_in[5];
  // only the last layer's weights matter (reference overwrites `out` each i)
  const float* qw  = (const float*)d_in[6] + (size_t)(NLAY_ - 1) * 768 * 768;
  const float* kvw = (const float*)d_in[7] + (size_t)(NLAY_ - 1) * 768 * 1536;
  const float* pw  = (const float*)d_in[8] + (size_t)(NLAY_ - 1) * 768 * 768;
  const float* pbv = (const float*)d_in[9] + (size_t)(NLAY_ - 1) * 768;
  float* out = (float*)d_out;

  char* ws = (char*)d_ws;
  float*    x0  = (float*)    (ws);               // 12,582,912 B
  _Float16* x0h = (_Float16*) (ws + 12582912);    //  6,291,456 B
  _Float16* wT  = (_Float16*) (ws + 18874368);    //  4,718,592 B
  _Float16* qsb = (_Float16*) (ws + 23592960);    //  6,291,456 B
  _Float16* ksb = (_Float16*) (ws + 29884416);    //  6,291,456 B
  _Float16* vtb = (_Float16*) (ws + 36175872);    //  6,291,456 B
  _Float16* ohb = x0h;  // x0h dead after qkv_gemm -> reuse for attention out

  wconv_kernel<<<dim3((3072 * 768 + 255) / 256), 256, 0, stream>>>(qw, kvw, pw, wT);
  patch_ln_kernel<<<dim3(4096), 256, 0, stream>>>(sam, convw, convb, lng, lnb,
                                                  pos, x0, x0h);
  qkv_gemm_kernel<<<dim3(128, 9), 128, 0, stream>>>(x0h, wT, qsb, ksb, vtb);
  attn_kernel<<<dim3(8, 8, 4), 128, 0, stream>>>(qsb, ksb, vtb, ohb);
  proj_kernel<<<dim3(128, 3), 128, 0, stream>>>(ohb, wT, pbv, x0, out);
}
